// WindowAttention_69982197121787
// MI455X (gfx1250) — matrix operand order repaired and checked by reading, not compiled, not run on hardware
//
#include <hip/hip_runtime.h>
#include <cstdint>
#include <cmath>

// CDNA5 / gfx1250: wave32, WMMA 16x16x32 bf16, fp32 accumulate.
typedef __attribute__((ext_vector_type(16))) __bf16 v16bf;
typedef __attribute__((ext_vector_type(2)))  __bf16 v2bf;
typedef __attribute__((ext_vector_type(8)))  float  v8f;

struct alignas(16) U4 { unsigned x, y, z, w; };
struct BV32 { U4 lo, hi; };                 // 32B = one v16bf operand
struct alignas(16) F4 { float x, y, z, w; };

#define LOG2E 1.4426950408889634f

__device__ __forceinline__ float fast_exp2(float x) {
#if __has_builtin(__builtin_amdgcn_exp2f)
  return __builtin_amdgcn_exp2f(x);         // v_exp_f32 directly
#else
  return exp2f(x);
#endif
}

__device__ __forceinline__ unsigned short f2bf(float f) {
  unsigned u = __builtin_bit_cast(unsigned, f);
  u += 0x7FFFu + ((u >> 16) & 1u);          // round-to-nearest-even
  return (unsigned short)(u >> 16);
}

// pack two f32 -> two bf16 in one dword (hardware v_cvt_pk_bf16_f32 when available)
__device__ __forceinline__ unsigned pk_bf16(float a, float b) {
#if __has_builtin(__builtin_amdgcn_cvt_pk_bf16_f32)
  v2bf r = __builtin_amdgcn_cvt_pk_bf16_f32(a, b);
  return __builtin_bit_cast(unsigned, r);
#else
  return (unsigned)f2bf(a) | ((unsigned)f2bf(b) << 16);
#endif
}

__device__ __forceinline__ v16bf mkv(U4 lo, U4 hi) {
  BV32 t; t.lo = lo; t.hi = hi;
  return __builtin_bit_cast(v16bf, t);
}

__device__ __forceinline__ v16bf ones_bf16() {   // B operand of all 1.0 (rowsum trick)
  U4 o; o.x = o.y = o.z = o.w = 0x3F803F80u;
  return mkv(o, o);
}

__device__ __forceinline__ v8f splat8(float v) {
  v8f r;
#pragma unroll
  for (int i = 0; i < 8; ++i) r[i] = v;
  return r;
}

// ---- WMMA operand loaders (contiguous k order K=16h+e, same in A and B) ----
// 16-bit A 16x32: lane l (m=l&15, h=l>>4): elem e = K 16h+e
__device__ __forceinline__ v16bf load_A(const unsigned short* p, int r0, int c0, int stride, int lane) {
  const unsigned short* row = p + (r0 + (lane & 15)) * stride + c0 + 16 * (lane >> 4);
  return mkv(*(const U4*)row, *(const U4*)(row + 8));
}
// A tile with only K=0..15 valid (QK^T, hd=16 zero-padded to 32)
__device__ __forceinline__ v16bf load_A_k16(const unsigned short* p, int r0, int c0, int stride, int lane) {
  const unsigned short* row = p + (r0 + (lane & 15)) * stride + c0;
  U4 z; z.x = z.y = z.z = z.w = 0u;
  return (lane < 16) ? mkv(*(const U4*)row, *(const U4*)(row + 8)) : mkv(z, z);
}
// 16-bit B 32x16: lane l (n=l&15, h=l>>4): elem e = B[K=16h+e][n]; source row-major with row=N, col=K
__device__ __forceinline__ v16bf load_B(const unsigned short* p, int r0, int c0, int stride, int lane) {
  const U4* q = (const U4*)(p + (r0 + (lane & 15)) * stride + c0 + 16 * (lane >> 4));
  return mkv(q[0], q[1]);
}
// B tile with K rows 16..31 = 0: lanes 16..31 read a 32B zeroed LDS stub (address select, not data select)
__device__ __forceinline__ v16bf load_B_k16(const unsigned short* p, int r0, int c0, int stride, int lane,
                                            const unsigned short* zero32) {
  const unsigned short* src = (lane < 16) ? (p + (r0 + (lane & 15)) * stride + c0) : zero32;
  const U4* q = (const U4*)src;
  return mkv(q[0], q[1]);
}
// pre-swizzled weight tile from L2: 32 lanes x 32B contiguous per 16x16x32 B tile
__device__ __forceinline__ v16bf load_B_sw(const unsigned short* w, int tile, int lane) {
  const U4* q = (const U4*)(w + tile * 512 + lane * 16);
  return mkv(q[0], q[1]);
}

#define WMMA_BF16(A, B, Cacc) \
  __builtin_amdgcn_wmma_f32_16x16x32_bf16(false, (A), false, (B), (short)0, (Cacc), false, false)

// ---------------- prep: swizzle weights into WMMA-B lane order (bf16) ----------------
// qkv tiles: 12 N-tiles x 2 K-steps = 24 tiles; Q tiles pre-scaled by hd^-0.5 * log2(e)
// (scores come out in log2 domain -> exp2 softmax with no per-element mul); proj: 8 tiles.
__global__ void prep_weights(const float* __restrict__ w_qkv, const float* __restrict__ w_proj,
                             unsigned short* __restrict__ ws) {
  int tid  = blockIdx.x * blockDim.x + threadIdx.x;   // 0..1023
  int tile = tid >> 5, lane = tid & 31;
  int nt = (tile < 24 ? tile : tile - 24) >> 1;
  int kt = tile & 1;
  int d  = nt * 16 + (lane & 15);                     // B column N = output dim
  int c0 = kt * 32 + 16 * (lane >> 4);                // B row K = input channel
  const float* w = (tile < 24) ? w_qkv : w_proj;      // [D][64], B[k][n] = w[n][k]
  const float scale = (tile < 8) ? (0.25f * LOG2E) : 1.0f;
  unsigned short* dst = ws + tile * 512 + lane * 16;
#pragma unroll
  for (int e = 0; e < 16; ++e) dst[e] = f2bf(w[d * 64 + c0 + e] * scale);
}

// ---------------- fused window attention: 1 block = 1 window, 1 wave = 1 head ----------------
__global__ __launch_bounds__(128) void win_attn(
    const float* __restrict__ vid, const float* __restrict__ modulator,
    const float* __restrict__ b_qkv, const float* __restrict__ b_proj,
    const unsigned short* __restrict__ wsw, float* __restrict__ out) {
  __shared__ unsigned short sX[64 * 64];        //  8 KB  x + modulator (bf16)
  __shared__ unsigned short sQKV[64 * 128 + 64 * 64]; // 24 KB: [0,8192)=Q|K row-major, [8192,..)=V^T
  __shared__ unsigned short sP[4 * 16 * 64];    //  8 KB  per-wave unnormalized probs
  __shared__ unsigned short sO[64 * 64];        //  8 KB  attention out [token][chan]
  __shared__ __align__(16) unsigned short sZero[16]; // 32B zero stub for K-padding

  const int b  = blockIdx.x;
  const int t  = b >> 12;                       // 64*64 windows per frame
  const int h0 = ((b >> 6) & 63) * 8;
  const int w0 = (b & 63) * 8;

  const int tid  = threadIdx.x;
  const int lane = tid & 31;
  const int wav  = tid >> 5;
  const int lh   = lane >> 4;
  const int ln   = lane & 15;

  // ---- load window (fp32 b128) + modulator -> bf16 LDS ----
  {
    if (tid < 16) sZero[tid] = 0;
    const int n = tid >> 1, half = tid & 1;     // token, channel-half
    const int py = n >> 3, px = n & 7;
    const F4* src = (const F4*)(vid + (((size_t)(t * 512 + h0 + py) * 512) + (w0 + px)) * 64 + half * 32);
    const F4* mod = (const F4*)(modulator + n * 64 + half * 32);
    unsigned short* dst = sX + n * 64 + half * 32;
#pragma unroll
    for (int j = 0; j < 8; ++j) {
      F4 a = src[j]; F4 m = mod[j];
      unsigned p0 = pk_bf16(a.x + m.x, a.y + m.y);
      unsigned p1 = pk_bf16(a.z + m.z, a.w + m.w);
      *(uint64_t*)(dst + j * 4) = (uint64_t)p0 | ((uint64_t)p1 << 32);
    }
  }
  __syncthreads();

  // ---- QKV = (x+mod) @ Wqkv^T + b ; 3 of 12 N-tiles per wave; branchless Q|K vs V^T stores ----
#pragma unroll
  for (int i = 0; i < 3; ++i) {
    const int nt   = wav * 3 + i;
    const int dcol = nt * 16 + ln;
    const bool isV = (nt >= 8);
    // bias splat materialized ONCE per N-tile; first WMMA of each chain consumes it as C
    const v8f biasv = splat8(b_qkv[dcol] * ((nt < 4) ? (0.25f * LOG2E) : 1.0f));
    const int vbase = 8192 + (dcol - 128) * 64;                  // V^T row base (valid only if isV)
    const v16bf B0 = load_B_sw(wsw, nt * 2 + 0, lane);
    const v16bf B1 = load_B_sw(wsw, nt * 2 + 1, lane);
#pragma unroll
    for (int mt = 0; mt < 4; ++mt) {
      v8f acc = WMMA_BF16(load_A(sX, mt * 16,  0, 64, lane), B0, biasv);
      acc     = WMMA_BF16(load_A(sX, mt * 16, 32, 64, lane), B1, acc);
#pragma unroll
      for (int j = 0; j < 4; ++j) {
        const int n0 = mt * 16 + 2 * j + 8 * lh;    // C-layout rows for acc[2j], acc[2j+1]
        const int n1 = n0 + 1;
        const unsigned p = pk_bf16(acc[2 * j], acc[2 * j + 1]);
        const int off0 = isV ? (vbase + n0) : (n0 * 128 + dcol);
        const int off1 = isV ? (vbase + n1) : (n1 * 128 + dcol);
        sQKV[off0] = (unsigned short)p;
        sQKV[off1] = (unsigned short)(p >> 16);
      }
    }
  }
  __syncthreads();

  // ---- attention: wave = head; scores in log2 domain ----
  {
    const int hh = wav;
    const unsigned short* sQK = sQKV;            // stride 128
    const unsigned short* sVt = sQKV + 8192;     // stride 64
    unsigned short* Pw = sP + wav * 16 * 64;
    // mt-invariant B operands, loaded once per head
    const v16bf bk0 = load_B_k16(sQK,  0, 64 + hh * 16, 128, lane, sZero);
    const v16bf bk1 = load_B_k16(sQK, 16, 64 + hh * 16, 128, lane, sZero);
    const v16bf bk2 = load_B_k16(sQK, 32, 64 + hh * 16, 128, lane, sZero);
    const v16bf bk3 = load_B_k16(sQK, 48, 64 + hh * 16, 128, lane, sZero);
    const v16bf bv0 = load_B(sVt, hh * 16,  0, 64, lane);
    const v16bf bv1 = load_B(sVt, hh * 16, 32, 64, lane);
    const v16bf bones = ones_bf16();
    const v8f zerov = splat8(0.0f);              // shared zero C operand
#pragma unroll 1
    for (int mt = 0; mt < 4; ++mt) {
      v16bf Aq = load_A_k16(sQK, mt * 16, hh * 16, 128, lane);
      v8f s0 = WMMA_BF16(Aq, bk0, zerov);
      v8f s1 = WMMA_BF16(Aq, bk1, zerov);
      v8f s2 = WMMA_BF16(Aq, bk2, zerov);
      v8f s3 = WMMA_BF16(Aq, bk3, zerov);
      // row max (cross-lane) + exp2; store UNNORMALIZED probs (rowsum comes from WMMA below)
#pragma unroll
      for (int r = 0; r < 8; ++r) {
        float m = fmaxf(fmaxf(s0[r], s1[r]), fmaxf(s2[r], s3[r]));
        m = fmaxf(m, __shfl_xor(m, 1)); m = fmaxf(m, __shfl_xor(m, 2));
        m = fmaxf(m, __shfl_xor(m, 4)); m = fmaxf(m, __shfl_xor(m, 8));
        s0[r] = fast_exp2(s0[r] - m); s1[r] = fast_exp2(s1[r] - m);
        s2[r] = fast_exp2(s2[r] - m); s3[r] = fast_exp2(s3[r] - m);
      }
      // C-layout -> row-major bf16 probs (A-operand staging), packed conversions
#pragma unroll
      for (int j = 0; j < 4; ++j) {
        const int row0 = 2 * j + 8 * lh, row1 = row0 + 1;
        unsigned q0 = pk_bf16(s0[2 * j], s0[2 * j + 1]);
        unsigned q1 = pk_bf16(s1[2 * j], s1[2 * j + 1]);
        unsigned q2 = pk_bf16(s2[2 * j], s2[2 * j + 1]);
        unsigned q3 = pk_bf16(s3[2 * j], s3[2 * j + 1]);
        Pw[row0 * 64 +  0 + ln] = (unsigned short)q0; Pw[row1 * 64 +  0 + ln] = (unsigned short)(q0 >> 16);
        Pw[row0 * 64 + 16 + ln] = (unsigned short)q1; Pw[row1 * 64 + 16 + ln] = (unsigned short)(q1 >> 16);
        Pw[row0 * 64 + 32 + ln] = (unsigned short)q2; Pw[row1 * 64 + 32 + ln] = (unsigned short)(q2 >> 16);
        Pw[row0 * 64 + 48 + ln] = (unsigned short)q3; Pw[row1 * 64 + 48 + ln] = (unsigned short)(q3 >> 16);
      }
      __asm__ volatile("s_wait_dscnt 0" ::: "memory");   // same-wave LDS RAW fence
      // O = P*V and rowsum = P*1 (shared A operands; per-lane sum lands in matching C register)
      const v16bf Ap0 = load_A(Pw, 0,  0, 64, lane);
      const v16bf Ap1 = load_A(Pw, 0, 32, 64, lane);
      v8f o    = WMMA_BF16(Ap0, bv0, zerov);
      o        = WMMA_BF16(Ap1, bv1, o);
      v8f ssum = WMMA_BF16(Ap0, bones, zerov);
      ssum     = WMMA_BF16(Ap1, bones, ssum);
#pragma unroll
      for (int j = 0; j < 4; ++j) {
        const int n0 = mt * 16 + 2 * j + 8 * lh;
        const float oa = o[2 * j]     * __frcp_rn(ssum[2 * j]);
        const float ob = o[2 * j + 1] * __frcp_rn(ssum[2 * j + 1]);
        const unsigned p = pk_bf16(oa, ob);
        sO[n0 * 64 + hh * 16 + ln]       = (unsigned short)p;
        sO[(n0 + 1) * 64 + hh * 16 + ln] = (unsigned short)(p >> 16);
      }
    }
  }
  __syncthreads();

  // ---- proj: out = O @ Wproj^T + b ; wave = M-tile; fp32 store to global ----
  {
    const int mt = wav;
#pragma unroll
    for (int nt = 0; nt < 4; ++nt) {
      const int cc = nt * 16 + ln;
      const v8f biasv = splat8(b_proj[cc]);
      v8f acc = WMMA_BF16(load_A(sO, mt * 16,  0, 64, lane),
                          load_B_sw(wsw + 24 * 512, nt * 2 + 0, lane), biasv);
      acc     = WMMA_BF16(load_A(sO, mt * 16, 32, 64, lane),
                          load_B_sw(wsw + 24 * 512, nt * 2 + 1, lane), acc);
#pragma unroll
      for (int r = 0; r < 8; ++r) {
        const int n = mt * 16 + r + 8 * lh;
        const int py = n >> 3, px = n & 7;
        out[(((size_t)(t * 512 + h0 + py) * 512) + (w0 + px)) * 64 + cc] = acc[r];
      }
    }
  }
}

extern "C" void kernel_launch(void* const* d_in, const int* in_sizes, int n_in,
                              void* d_out, int out_size, void* d_ws, size_t ws_size,
                              hipStream_t stream) {
  const float* vid       = (const float*)d_in[0];
  const float* modulator = (const float*)d_in[1];
  const float* w_qkv     = (const float*)d_in[2];
  const float* b_qkv     = (const float*)d_in[3];
  const float* w_proj    = (const float*)d_in[4];
  const float* b_proj    = (const float*)d_in[5];
  float* out = (float*)d_out;
  unsigned short* wsw = (unsigned short*)d_ws;   // 32 KB: 24+8 swizzled bf16 B-tiles

  prep_weights<<<dim3(8), dim3(128), 0, stream>>>(w_qkv, w_proj, wsw);
  win_attn<<<dim3(16384), dim3(128), 0, stream>>>(vid, modulator, b_qkv, b_proj, wsw, out);
}
